// NovelNet_14740327760514
// MI455X (gfx1250) — matrix operand order repaired and checked by reading, not compiled, not run on hardware
//
#include <hip/hip_runtime.h>
#include <cmath>
#include <cstdint>

// ---------------- problem constants ----------------
#define BB   256
#define SS   50
#define NV   100000
#define DEe  128
#define DFe  32
#define HD   64
#define GINe 352          // 128 + 7*32
#define NGATE 192         // 3*HD
#define NEGV (-1.0e9f)
#define VNT  5            // vocab n-tiles per block (6250 = 1250*5)

typedef __attribute__((ext_vector_type(16))) _Float16 v16h;
typedef __attribute__((ext_vector_type(8)))  _Float16 v8h;
typedef __attribute__((ext_vector_type(8)))  float    v8f;
typedef __attribute__((ext_vector_type(4)))  unsigned int v4u;
typedef __attribute__((ext_vector_type(8)))  int v8i;
typedef __attribute__((ext_vector_type(4)))  int v4i;

// ---------------- WMMA fragment loaders (CDNA5 16x16x32 f16 layouts) -------
// A: 16x32 f16, row-major source with leading dim lda (halves).
__device__ inline v16h ldA(const _Float16* A, int lda, int m0, int k0, int lane) {
    int m  = m0 + (lane & 15);
    int kb = k0 + ((lane >> 4) << 3);
    const _Float16* p = A + (long)m * lda + kb;
    v8h lo = *(const v8h*)(p);
    v8h hi = *(const v8h*)(p + 16);
    v16h r;
#pragma unroll
    for (int i = 0; i < 8; ++i) { r[i] = lo[i]; r[i + 8] = hi[i]; }
    return r;
}
// B: 32x16 f16 from row-major W (N x K): lane n=l&15 takes K in the same order as ldA.
__device__ inline v16h ldB(const _Float16* W, int ldb, int n0, int k0, int lane) {
    int n  = n0 + (lane & 15);
    int kb = k0 + ((lane >> 4) << 3);
    const _Float16* p = W + (long)n * ldb + kb;
    v8h lo = *(const v8h*)(p);
    v8h hi = *(const v8h*)(p + 16);
    v16h r;
#pragma unroll
    for (int i = 0; i < 8; ++i) { r[i] = lo[i]; r[i + 8] = hi[i]; }
    return r;
}

__device__ inline void atomicMaxF(float* addr, float val) {
    unsigned int* ua  = (unsigned int*)addr;
    unsigned int  old = *ua;
    while (__uint_as_float(old) < val) {
        unsigned int assumed = old;
        old = atomicCAS(ua, assumed, __float_as_uint(val));
        if (old == assumed) break;
    }
}

// ---------------- small utility kernels ----------------
__global__ void k_f32_to_f16(const float* __restrict__ src, _Float16* __restrict__ dst, long n) {
    long i = (long)blockIdx.x * blockDim.x + threadIdx.x;
    if (i < n) dst[i] = (_Float16)src[i];
}
__global__ void k_f32copy(const float* __restrict__ src, float* __restrict__ dst, long n) {
    long i = (long)blockIdx.x * blockDim.x + threadIdx.x;
    if (i < n) dst[i] = src[i];
}
__global__ void k_memset_u8(unsigned char* __restrict__ p, long n) {
    long i = (long)blockIdx.x * blockDim.x + threadIdx.x;
    if (i < n) p[i] = 0;
}
__global__ void k_mask_scatter(const int* __restrict__ ids, unsigned char* __restrict__ msk) {
    int i = blockIdx.x * blockDim.x + threadIdx.x;
    if (i < BB * SS) {
        int id = ids[i];
        if (id > 0) msk[(long)(i / SS) * NV + id] = 1;
    }
}
__global__ void k_init_rows(float* __restrict__ rowmax, float* __restrict__ rowsum) {
    int i = threadIdx.x;   // 256 threads
    rowmax[i] = -3.0e38f;
    rowsum[i] = 0.0f;
}

// ---------------- embedding gather -> seq f16 (B*S x 352) ----------------
__global__ void k_gather(const int* __restrict__ ids,
                         const int* f0, const int* f1, const int* f2, const int* f3,
                         const int* f4, const int* f5, const int* f6,
                         const float* __restrict__ nemb,
                         const float* t0, const float* t1, const float* t2, const float* t3,
                         const float* t4, const float* t5, const float* t6,
                         _Float16* __restrict__ seq) {
    int row = blockIdx.x;           // b*S+s
    int col = threadIdx.x;          // 0..351
    const int*   feats[7] = {f0, f1, f2, f3, f4, f5, f6};
    const float* tabs[7]  = {t0, t1, t2, t3, t4, t5, t6};
    float v;
    if (col < DEe) {
        v = nemb[(long)ids[row] * DEe + col];
    } else {
        int fi = (col - DEe) >> 5;
        int c  = (col - DEe) & 31;
        v = tabs[fi][(long)feats[fi][row] * DFe + c];
    }
    seq[(long)row * GINe + col] = (_Float16)v;
}

// ---------------- generic WMMA GEMM: C = A(MxK) * W(NxK)^T (+bias) --------
__global__ void k_gemm_wmma(const _Float16* __restrict__ A, const _Float16* __restrict__ W,
                            const float* __restrict__ bias, float* __restrict__ C,
                            int M, int N, int K, int lda, int ldb, int ldc) {
    const int lane = threadIdx.x & 31;
    const int wave = threadIdx.x >> 5;
    const int tile = blockIdx.x * (blockDim.x >> 5) + wave;
    const int ntiles = N >> 4;
    const int mt = tile / ntiles;
    const int nt = tile - mt * ntiles;
    if (mt * 16 >= M) return;
    v8f acc = {};
    for (int k0 = 0; k0 < K; k0 += 32) {
        v16h a = ldA(A, lda, mt * 16, k0, lane);
        v16h b = ldB(W, ldb, nt * 16, k0, lane);
        acc = __builtin_amdgcn_wmma_f32_16x16x32_f16(false, a, false, b, (short)0, acc,
                                                     false, false);
    }
    const int n  = nt * 16 + (lane & 15);
    const int mb = mt * 16 + ((lane >> 4) << 3);
    const float bv = bias ? bias[n] : 0.0f;
#pragma unroll
    for (int r = 0; r < 8; ++r)
        C[(long)(mb + r) * ldc + n] = acc[r] + bv;
}

// ---------------- bidirectional GRU recurrence ----------------
__global__ void k_gru(const float* __restrict__ gi,          // (B*S, 384) f32, bih folded
                      const _Float16* __restrict__ whh16,    // (384, 64) f16 packed f|b
                      const float* __restrict__ bhh,         // (384) packed f|b
                      _Float16* __restrict__ enc16,          // (B, S, 128)
                      _Float16* __restrict__ binfo16) {      // (B, 128)
    __shared__ __align__(32) _Float16 hh[16 * HD];
    __shared__ float hf[16 * HD];
    __shared__ float gh[16 * NGATE];
    const int lane = threadIdx.x;                    // 32 threads
    const int dir  = blockIdx.x & 1;
    const int bt   = blockIdx.x >> 1;
    for (int e = lane; e < 16 * HD; e += 32) { hh[e] = (_Float16)0.0f; hf[e] = 0.0f; }
    __syncthreads();
    const _Float16* Wb = whh16 + (long)dir * NGATE * HD;
    const float*    bh = bhh + dir * NGATE;
    for (int step = 0; step < SS; ++step) {
        const int s = dir ? (SS - 1 - step) : step;
        for (int nt = 0; nt < NGATE / 16; ++nt) {
            v8f acc = {};
#pragma unroll
            for (int k0 = 0; k0 < HD; k0 += 32) {
                v16h a = ldA(hh, HD, 0, k0, lane);
                v16h b = ldB(Wb, HD, nt * 16, k0, lane);
                acc = __builtin_amdgcn_wmma_f32_16x16x32_f16(false, a, false, b, (short)0,
                                                             acc, false, false);
            }
            const int n  = nt * 16 + (lane & 15);
            const int mb = (lane >> 4) << 3;
#pragma unroll
            for (int r = 0; r < 8; ++r) gh[(mb + r) * NGATE + n] = acc[r];
        }
        __syncthreads();
        for (int e = lane; e < 16 * HD; e += 32) {
            const int i = e >> 6, j = e & 63;
            const int b = bt * 16 + i;
            const long gr = ((long)b * SS + s) * (2 * NGATE) + (long)dir * NGATE;
            float gir = gi[gr + j], giz = gi[gr + 64 + j], gin = gi[gr + 128 + j];
            float ghr = gh[i * NGATE + j]       + bh[j];
            float ghz = gh[i * NGATE + 64 + j]  + bh[64 + j];
            float ghn = gh[i * NGATE + 128 + j] + bh[128 + j];
            float r = 1.0f / (1.0f + expf(-(gir + ghr)));
            float z = 1.0f / (1.0f + expf(-(giz + ghz)));
            float nn = tanhf(gin + r * ghn);
            float hnew = (1.0f - z) * nn + z * hf[e];
            hf[e] = hnew;
            hh[e] = (_Float16)hnew;
            enc16[((long)b * SS + s) * 128 + dir * HD + j] = (_Float16)hnew;
        }
        __syncthreads();
    }
    for (int e = lane; e < 16 * HD; e += 32) {
        const int i = e >> 6, j = e & 63;
        binfo16[(long)(bt * 16 + i) * 128 + dir * HD + j] = hh[e];
    }
}

// ---------------- attention scores (both heads) ----------------
__global__ void k_attn_sc(const float* __restrict__ t1, const float* __restrict__ t2,
                          const float* __restrict__ vNew, const float* __restrict__ vCon,
                          const int* __restrict__ ids, const int* __restrict__ lastm,
                          float* __restrict__ scN, float* __restrict__ scC) {
    const int row = blockIdx.x;           // b*S+s
    const int lane = threadIdx.x;         // 32
    const int b = row / SS;
    float aN = 0.0f, aC = 0.0f;
    for (int h = lane; h < 128; h += 32) {
        float uN = tanhf(t1[(long)row * 256 + h]       + t2[(long)b * 256 + h]);
        float uC = tanhf(t1[(long)row * 256 + 128 + h] + t2[(long)b * 256 + 128 + h]);
        aN += vNew[h] * uN;
        aC += vCon[h] * uC;
    }
#pragma unroll
    for (int off = 16; off; off >>= 1) {
        aN += __shfl_down(aN, off, 32);
        aC += __shfl_down(aC, off, 32);
    }
    if (lane == 0) {
        scN[row] = (ids[row] == 0) ? NEGV : aN;
        scC[row] = (lastm[row] == 0) ? NEGV : aC;
    }
}

// ---------------- softmax over S, rseq, a_con, a_con_sig ----------------
__global__ void k_attn_finalize(const float* __restrict__ scN, const float* __restrict__ scC,
                                const _Float16* __restrict__ enc16,
                                float* __restrict__ aCon, float* __restrict__ outSig,
                                _Float16* __restrict__ rseq16) {
    const int b = blockIdx.x;
    const int lane = threadIdx.x;         // 32
    __shared__ float an[64];
    float x0 = scN[(long)b * SS + lane];
    float x1 = (lane + 32 < SS) ? scN[(long)b * SS + lane + 32] : NEGV;
    float mx = fmaxf(x0, x1);
#pragma unroll
    for (int off = 16; off; off >>= 1) mx = fmaxf(mx, __shfl_xor(mx, off, 32));
    float e0 = expf(x0 - mx);
    float e1 = (lane + 32 < SS) ? expf(x1 - mx) : 0.0f;
    float sm = e0 + e1;
#pragma unroll
    for (int off = 16; off; off >>= 1) sm += __shfl_xor(sm, off, 32);
    an[lane] = e0 / sm;
    if (lane + 32 < SS) an[lane + 32] = e1 / sm;
    __syncthreads();
    for (int h = lane; h < 128; h += 32) {
        float acc = 0.0f;
        for (int s = 0; s < SS; ++s)
            acc += an[s] * (float)enc16[((long)b * SS + s) * 128 + h];
        rseq16[(long)b * 128 + h] = (_Float16)acc;
    }
    float y0 = scC[(long)b * SS + lane];
    float y1 = (lane + 32 < SS) ? scC[(long)b * SS + lane + 32] : NEGV;
    float mc = fmaxf(y0, y1);
#pragma unroll
    for (int off = 16; off; off >>= 1) mc = fmaxf(mc, __shfl_xor(mc, off, 32));
    float c0 = expf(y0 - mc);
    float c1 = (lane + 32 < SS) ? expf(y1 - mc) : 0.0f;
    float sc = c0 + c1;
#pragma unroll
    for (int off = 16; off; off >>= 1) sc += __shfl_xor(sc, off, 32);
    aCon[(long)b * SS + lane]   = c0 / sc;
    outSig[(long)b * SS + lane] = 1.0f / (1.0f + expf(-y0));
    if (lane + 32 < SS) {
        aCon[(long)b * SS + lane + 32]   = c1 / sc;
        outSig[(long)b * SS + lane + 32] = 1.0f / (1.0f + expf(-y1));
    }
}

// ---------------- vocab GEMM + online softmax passes ----------------
// Block = 16 waves (one per m-tile of M=256) x VNT n-tiles.
// B tile (VNT*16 vocab rows, 20 KB) staged into LDS once per block via the
// Tensor Data Mover; WMMA B-fragments then come from ds_load.
// mode 0: row max (atomic), 1: sum exp (atomic), 2: write prob
__global__ void __launch_bounds__(512)
k_vocab_pass(const _Float16* __restrict__ rseqh, const _Float16* __restrict__ embh,
             const unsigned char* __restrict__ msk,
             float* __restrict__ rowmax, float* __restrict__ rowsum,
             float* __restrict__ prob, int mode) {
    __shared__ __align__(32) _Float16 Bs[VNT * 16 * 128];   // 20 KB
    const int lane = threadIdx.x & 31;
    const int wv   = threadIdx.x >> 5;          // 0..15 == m-tile
    const int n0   = blockIdx.x * (VNT * 16);
    const _Float16* gsrc = embh + (long)n0 * 128;

#if __has_builtin(__builtin_amdgcn_tensor_load_to_lds) && \
    __has_builtin(__builtin_amdgcn_s_wait_tensorcnt)
    if (wv == 0) {
        // Tensor DMA descriptor (D#): flat 1-D tile of 10240 f16 elements.
        unsigned long long ga = (unsigned long long)(uintptr_t)gsrc;
        unsigned int ldsa = (unsigned int)(uintptr_t)(void*)&Bs[0];
        const unsigned int nelem = VNT * 16 * 128;           // 10240
        v4u g0;
        g0[0] = 1u;                                          // count=1 (valid, user)
        g0[1] = ldsa;                                        // lds_addr
        g0[2] = (unsigned int)(ga & 0xffffffffu);            // global_addr[31:0]
        g0[3] = (unsigned int)((ga >> 32) & 0x01ffffffu) | (2u << 30);  // addr[56:32]|type=2
        v8i g1;
        g1[0] = (int)(1u << 16);                             // data_size=2B, no flags
        g1[1] = (int)((nelem & 0xffffu) << 16);              // tensor_dim0[15:0]
        g1[2] = (int)(((nelem >> 16) & 0xffffu) | (1u << 16)); // dim0[31:16] | tensor_dim1=1
        g1[3] = (int)((nelem & 0xffffu) << 16);              // tile_dim0 = nelem
        g1[4] = 1;                                           // tile_dim1 = 1
        g1[5] = (int)nelem;                                  // tensor_dim0_stride lo
        g1[6] = 0;
        g1[7] = 0;
        v4i gz4  = {0, 0, 0, 0};                             // D# groups 2/3 unused (2-D tile)
        v8i gz8  = {0, 0, 0, 0, 0, 0, 0, 0};
        __builtin_amdgcn_tensor_load_to_lds(g0, g1, gz4, gz4, gz8, 0);
        __builtin_amdgcn_s_wait_tensorcnt(0);
    }
#else
    for (int e = threadIdx.x; e < VNT * 16 * 128 / 8; e += (int)blockDim.x)
        ((v8h*)Bs)[e] = ((const v8h*)gsrc)[e];
#endif
    __syncthreads();

    if (threadIdx.x == 0 && n0 + VNT * 16 < NV)   // warm next block's B region in L2
        __builtin_prefetch((const void*)(embh + ((long)n0 + VNT * 16) * 128), 0, 1);

    // A fragments for this wave's m-tile, held in registers across all n-tiles.
    v16h afrag[4];
#pragma unroll
    for (int kk = 0; kk < 4; ++kk) afrag[kk] = ldA(rseqh, 128, wv * 16, kk * 32, lane);

    const int mb = wv * 16 + ((lane >> 4) << 3);
#pragma unroll
    for (int t = 0; t < VNT; ++t) {
        v8f acc = {};
#pragma unroll
        for (int kk = 0; kk < 4; ++kk) {
            v16h b = ldB(Bs, 128, t * 16, kk * 32, lane);
            acc = __builtin_amdgcn_wmma_f32_16x16x32_f16(false, afrag[kk], false, b, (short)0,
                                                         acc, false, false);
        }
        const int n = n0 + t * 16 + (lane & 15);
        if (mode == 0) {
#pragma unroll
            for (int r = 0; r < 8; ++r) {
                const int m = mb + r;
                float v = msk[(long)m * NV + n] ? NEGV : acc[r];
#pragma unroll
                for (int off = 1; off < 16; off <<= 1) v = fmaxf(v, __shfl_xor(v, off, 32));
                if ((lane & 15) == 0) atomicMaxF(&rowmax[m], v);
            }
        } else if (mode == 1) {
#pragma unroll
            for (int r = 0; r < 8; ++r) {
                const int m = mb + r;
                float e = msk[(long)m * NV + n] ? 0.0f : expf(acc[r] - rowmax[m]);
#pragma unroll
                for (int off = 1; off < 16; off <<= 1) e += __shfl_xor(e, off, 32);
                if ((lane & 15) == 0) atomicAdd(&rowsum[m], e);
            }
        } else {
#pragma unroll
            for (int r = 0; r < 8; ++r) {
                const int m = mb + r;
                float p = msk[(long)m * NV + n] ? 0.0f : expf(acc[r] - rowmax[m]) / rowsum[m];
                prob[(long)m * NV + n] = p;
            }
        }
    }
}

__global__ void k_con_scatter(const int* __restrict__ ids, const float* __restrict__ aCon,
                              float* __restrict__ prob) {
    int i = blockIdx.x * blockDim.x + threadIdx.x;
    if (i < BB * SS)
        atomicAdd(&prob[(long)(i / SS) * NV + ids[i]], aCon[i]);
}

// ================= host side =================
extern "C" void kernel_launch(void* const* d_in, const int* in_sizes, int n_in,
                              void* d_out, int out_size, void* d_ws, size_t ws_size,
                              hipStream_t stream) {
    const int*   ids   = (const int*)d_in[0];
    const int*   f[7]  = {(const int*)d_in[1], (const int*)d_in[2], (const int*)d_in[3],
                          (const int*)d_in[4], (const int*)d_in[5], (const int*)d_in[6],
                          (const int*)d_in[7]};
    const int*   lastm = (const int*)d_in[8];
    const float* nemb  = (const float*)d_in[9];
    const float* tab[7] = {(const float*)d_in[10], (const float*)d_in[11], (const float*)d_in[12],
                           (const float*)d_in[13], (const float*)d_in[14], (const float*)d_in[15],
                           (const float*)d_in[16]};
    const float* wih_f = (const float*)d_in[17];
    const float* whh_f = (const float*)d_in[18];
    const float* bih_f = (const float*)d_in[19];
    const float* bhh_f = (const float*)d_in[20];
    const float* wih_b = (const float*)d_in[21];
    const float* whh_b = (const float*)d_in[22];
    const float* bih_b = (const float*)d_in[23];
    const float* bhh_b = (const float*)d_in[24];
    const float* newWh = (const float*)d_in[25];
    const float* newWs = (const float*)d_in[26];
    const float* newV  = (const float*)d_in[27];
    const float* conWh = (const float*)d_in[28];
    const float* conWs = (const float*)d_in[29];
    const float* conV  = (const float*)d_in[30];

    float* prob   = (float*)d_out;
    float* outSig = (float*)d_out + (size_t)BB * NV;

    char*  base = (char*)d_ws;
    size_t off  = 0;
    auto alloc = [&](size_t bytes) {
        size_t o = off;
        off = (off + bytes + 255) & ~(size_t)255;
        return o;
    };
    const long nRows = (long)BB * SS;                    // 12800
    _Float16* seq16   = (_Float16*)(base + alloc(nRows * GINe * 2));
    _Float16* Wih16   = (_Float16*)(base + alloc(2L * NGATE * GINe * 2));
    float*    bihP    = (float*)   (base + alloc(2L * NGATE * 4));
    float*    gi      = (float*)   (base + alloc(nRows * 2L * NGATE * 4));
    _Float16* Whh16   = (_Float16*)(base + alloc(2L * NGATE * HD * 2));
    float*    bhhP    = (float*)   (base + alloc(2L * NGATE * 4));
    _Float16* enc16   = (_Float16*)(base + alloc(nRows * 128L * 2));
    _Float16* binfo16 = (_Float16*)(base + alloc((long)BB * 128 * 2));
    _Float16* Wh16    = (_Float16*)(base + alloc(256L * 128 * 2));
    _Float16* Ws16    = (_Float16*)(base + alloc(256L * 128 * 2));
    float*    t1      = (float*)   (base + alloc(nRows * 256L * 4));
    float*    t2      = (float*)   (base + alloc((long)BB * 256 * 4));
    float*    scN     = (float*)   (base + alloc(nRows * 4));
    float*    scC     = (float*)   (base + alloc(nRows * 4));
    float*    aCon    = (float*)   (base + alloc(nRows * 4));
    _Float16* rseq16  = (_Float16*)(base + alloc((long)BB * 128 * 2));
    _Float16* emb16   = (_Float16*)(base + alloc((long)NV * DEe * 2));
    unsigned char* msk = (unsigned char*)(base + alloc((long)BB * NV));
    float*    rowmax  = (float*)   (base + alloc(BB * 4));
    float*    rowsum  = (float*)   (base + alloc(BB * 4));
    (void)ws_size; (void)n_in; (void)in_sizes; (void)out_size;

    auto cvt = [&](const float* s, _Float16* d, long n) {
        k_f32_to_f16<<<dim3((unsigned)((n + 255) / 256)), dim3(256), 0, stream>>>(s, d, n);
    };
    auto cpy = [&](const float* s, float* d, long n) {
        k_f32copy<<<dim3((unsigned)((n + 255) / 256)), dim3(256), 0, stream>>>(s, d, n);
    };

    cvt(wih_f, Wih16,                      (long)NGATE * GINe);
    cvt(wih_b, Wih16 + (long)NGATE * GINe, (long)NGATE * GINe);
    cvt(whh_f, Whh16,                      (long)NGATE * HD);
    cvt(whh_b, Whh16 + (long)NGATE * HD,   (long)NGATE * HD);
    cpy(bih_f, bihP, NGATE);  cpy(bih_b, bihP + NGATE, NGATE);
    cpy(bhh_f, bhhP, NGATE);  cpy(bhh_b, bhhP + NGATE, NGATE);
    cvt(newWh, Wh16, 128L * 128);  cvt(conWh, Wh16 + 128L * 128, 128L * 128);
    cvt(newWs, Ws16, 128L * 128);  cvt(conWs, Ws16 + 128L * 128, 128L * 128);
    cvt(nemb,  emb16, (long)NV * DEe);

    k_gather<<<dim3((unsigned)nRows), dim3(GINe), 0, stream>>>(
        ids, f[0], f[1], f[2], f[3], f[4], f[5], f[6],
        nemb, tab[0], tab[1], tab[2], tab[3], tab[4], tab[5], tab[6], seq16);

    {   // gi = seq @ Wih^T + bih
        int tiles = (int)(nRows / 16) * (2 * NGATE / 16);
        k_gemm_wmma<<<dim3(tiles / 4), dim3(128), 0, stream>>>(
            seq16, Wih16, bihP, gi, (int)nRows, 2 * NGATE, GINe, GINe, GINe, 2 * NGATE);
    }

    k_gru<<<dim3(2 * (BB / 16)), dim3(32), 0, stream>>>(gi, Whh16, bhhP, enc16, binfo16);

    {   // t1 = enc @ [Wh_new|Wh_con]^T
        int tiles = (int)(nRows / 16) * (256 / 16);
        k_gemm_wmma<<<dim3(tiles / 4), dim3(128), 0, stream>>>(
            enc16, Wh16, nullptr, t1, (int)nRows, 256, 128, 128, 128, 256);
    }
    {   // t2 = binfo @ [Ws_new|Ws_con]^T
        int tiles = (BB / 16) * (256 / 16);
        k_gemm_wmma<<<dim3(tiles / 4), dim3(128), 0, stream>>>(
            binfo16, Ws16, nullptr, t2, BB, 256, 128, 128, 128, 256);
    }

    k_attn_sc<<<dim3((unsigned)nRows), dim3(32), 0, stream>>>(
        t1, t2, newV, conV, ids, lastm, scN, scC);
    k_attn_finalize<<<dim3(BB), dim3(32), 0, stream>>>(
        scN, scC, enc16, aCon, outSig, rseq16);

    k_memset_u8<<<dim3((unsigned)(((long)BB * NV + 255) / 256)), dim3(256), 0, stream>>>(
        msk, (long)BB * NV);
    k_mask_scatter<<<dim3((unsigned)((nRows + 255) / 256)), dim3(256), 0, stream>>>(ids, msk);
    k_init_rows<<<dim3(1), dim3(BB), 0, stream>>>(rowmax, rowsum);

    {   // vocab GEMM 256 x 100000 x 128 with recompute-softmax (3 passes)
        dim3 g((NV / 16) / VNT), blk(512);                  // 1250 blocks
        k_vocab_pass<<<g, blk, 0, stream>>>(rseq16, emb16, msk, rowmax, rowsum, prob, 0);
        k_vocab_pass<<<g, blk, 0, stream>>>(rseq16, emb16, msk, rowmax, rowsum, prob, 1);
        k_vocab_pass<<<g, blk, 0, stream>>>(rseq16, emb16, msk, rowmax, rowsum, prob, 2);
    }

    k_con_scatter<<<dim3((unsigned)((nRows + 255) / 256)), dim3(256), 0, stream>>>(
        ids, aCon, prob);
}
